// ShiftBlock1_41609643163682
// MI455X (gfx1250) — 8-bit matrix operand order repaired and checked by reading, compiled, not run on hardware
//
#include <hip/hip_runtime.h>

typedef int v8i __attribute__((ext_vector_type(8)));

#define BT   12
#define C    64
#define H    256
#define W    256
#define TFR  3
#define SHIFT 32
#define QMOD 192

__global__ void prep_bin_weights(const float* __restrict__ w,
                                 float* __restrict__ scale,
                                 signed char* __restrict__ bpack) {
    int o = blockIdx.x;
    int i = threadIdx.x;

    float wv[9];
    float s = 0.f;
    int base = (o * 64 + i) * 9;
    #pragma unroll
    for (int k = 0; k < 9; ++k) {
        wv[k] = w[base + k];
        s += fabsf(wv[k]);
    }

    __shared__ float red[64];
    red[i] = s;
    __syncthreads();
    for (int off = 32; off > 0; off >>= 1) {
        if (i < off) red[i] += red[i + off];
        __syncthreads();
    }
    if (i == 0) scale[o] = red[0] / 576.0f;

    int lane  = (o & 15) | (((i >> 4) & 1) << 4);
    int byt   = (i & 15) | (((i >> 5) & 1) << 4);
    int ntile = o >> 4;
    #pragma unroll
    for (int k = 0; k < 9; ++k) {
        signed char sg = (signed char)((wv[k] > 0.f) - (wv[k] < 0.f));
        bpack[(size_t)((k * 4 + ntile) * 32 + lane) * 32 + byt] = sg;
    }
}

__global__ __launch_bounds__(256)
void binconv_shift_wmma(const float* __restrict__ x,
                        const float* __restrict__ scale,
                        const signed char* __restrict__ bpack,
                        float* __restrict__ out) {
    __shared__ __align__(16) unsigned char smem[64 * 33 * 4];
    signed char* sIn  = (signed char*)smem;
    float*       sOut = (float*)smem;

    const int tid  = threadIdx.x;
    const int lane = tid & 31;
    const int wave = tid >> 5;
    const int w0   = blockIdx.x * 32;
    const int h    = blockIdx.y;
    const int n    = blockIdx.z;
    const int bgrp = n / TFR;
    const int tfrm = n - bgrp * TFR;

    for (int it = 0; it < 26; ++it) {
        int e = it * 256 + tid;
        if (e < 34 * 3 * 64) {
            int col  = e % 34;
            int rest = e / 34;
            int row  = rest % 3;
            int ch   = rest / 3;
            int q      = tfrm * 64 + ch;
            int qs     = (q + (QMOD - SHIFT)) % QMOD;
            int src_bt = bgrp * TFR + (qs >> 6);
            int src_c  = qs & 63;
            int gh = h  + row - 1;
            int gw = w0 + col - 1;
            float v = 0.f;
            if ((unsigned)gh < H && (unsigned)gw < W) {
                v = x[(((size_t)src_bt * C + src_c) * H + gh) * W + gw];
            }
            signed char sg = (signed char)((v > 0.f) - (v < 0.f));
            sIn[(row * 34 + col) * 68 + ch] = sg;
        }
    }
    __syncthreads();

    const int mt = wave >> 2;
    const int nt = wave & 3;
    const int hi = (lane >= 16) ? 8 : 0;
    const int p  = mt * 16 + (lane & 15);

    v8i acc = {0, 0, 0, 0, 0, 0, 0, 0};

    #pragma unroll
    for (int kpos = 0; kpos < 9; ++kpos) {
        const int kh = kpos / 3;
        const int kw = kpos - kh * 3;

        const v8i bfrag =
            *(const v8i*)(bpack + (size_t)(((kpos * 4 + nt) * 32) + lane) * 32);

        const signed char* arow = sIn + (kh * 34 + (p + kw)) * 68 + 2 * hi;
        v8i afrag;
        #pragma unroll
        for (int v = 0; v < 8; ++v) {
            int kbase = ((v >> 2) << 5) + ((v & 3) << 2);
            afrag[v] = *(const int*)(arow + kbase);
        }

        acc = __builtin_amdgcn_wmma_i32_16x16x64_iu8(
             true, afrag,  true, bfrag, acc,
             false,  false);
    }

    const int oc = nt * 16 + (lane & 15);
    const float sc = scale[oc];

    __syncthreads();
    #pragma unroll
    for (int r = 0; r < 8; ++r) {
        int m  = r + hi;
        int px = mt * 16 + m;
        sOut[oc * 33 + px] = sc * (float)acc[r];
    }
    __syncthreads();

    const size_t obase = ((size_t)n * C) * (H * W) + (size_t)h * W + w0;
    #pragma unroll
    for (int j = 0; j < 8; ++j) {
        int f  = j * 256 + tid;
        int o2 = f >> 5;
        int p2 = f & 31;
        out[obase + (size_t)o2 * (H * W) + p2] = sOut[o2 * 33 + p2];
    }
}

extern "C" void kernel_launch(void* const* d_in, const int* in_sizes, int n_in,
                              void* d_out, int out_size, void* d_ws, size_t ws_size,
                              hipStream_t stream) {
    const float* x = (const float*)d_in[0];
    const float* w = (const float*)d_in[1];
    float* out = (float*)d_out;

    float* scale       = (float*)d_ws;
    signed char* bpack = (signed char*)d_ws + 256;

    prep_bin_weights<<<64, 64, 0, stream>>>(w, scale, bpack);

    dim3 grid(W / 32, H, BT);
    binconv_shift_wmma<<<grid, 256, 0, stream>>>(x, scale, bpack, out);
}
